// GCNProt_10024453669560
// MI455X (gfx1250) — matrix operand order repaired and checked by reading, not compiled, not run on hardware
//
#include <hip/hip_runtime.h>
#include <hip/hip_bf16.h>

#define NNODES 200000
#define NEDGES 6400000
#define FEAT   128
#define HID    96
#define NGRAPH 2048

typedef __bf16 bf16_t;
typedef __attribute__((ext_vector_type(16))) __bf16 v16bf;
typedef __attribute__((ext_vector_type(8)))  __bf16 v8bf;   // 16 bytes
typedef __attribute__((ext_vector_type(8)))  float  v8f;

// float -> bf16, round-to-nearest-even (pure integer ops; no libcall risk)
__device__ __forceinline__ bf16_t f2bf(float f) {
    union { float f; unsigned u; } v; v.f = f;
    unsigned r = v.u + 0x7FFFu + ((v.u >> 16) & 1u);
    unsigned short hs = (unsigned short)(r >> 16);
    union { unsigned short s; bf16_t b; } o; o.s = hs; return o.b;
}

// ---------------------------------------------------------------------------
// WMMA GEMM: C[nrows x 96] = A[nrows x K] @ W[K x 96]
// A: bf16 row-major in global.  W: fp32 in global, transposed into LDS as bf16.
// One wave computes a full 16x96 row-stripe (6 accumulators, 6 WMMA / k-step).
// A fragment  = two 16B global_load_b128 per k-step (contiguous K runs).
// B fragment  = one 32B LDS read (two ds_load_b128) per tile per k-step.
// nrows % 16 == 0, K % 32 == 0.
// ---------------------------------------------------------------------------
__global__ void gemm_wmma_bf16(const bf16_t* __restrict__ A,
                               const float* __restrict__ W,
                               float* __restrict__ C,
                               int nrows, int K) {
    __shared__ bf16_t Wt[HID * FEAT];            // W^T, max 96*128*2B = 24.6 KB
    const int tot = K * HID;
    for (int i = threadIdx.x; i < tot; i += blockDim.x) {
        int n = i / K, k = i - n * K;
        Wt[i] = f2bf(W[k * HID + n]);            // Wt[n*K + k] = W[k][n]
    }
    __syncthreads();

    const long long wave = ((long long)blockIdx.x * blockDim.x + threadIdx.x) >> 5;
    if (wave >= (long long)(nrows >> 4)) return;             // wave-uniform

    const int lane = threadIdx.x & 31;
    const int grp  = lane >> 4;                               // 0 / 1
    const int mr   = lane & 15;
    const int m0   = (int)wave << 4;

    const bf16_t* arow = A + (long long)(m0 + mr) * K;
    v8f acc[6] = {};

    for (int k0 = 0; k0 < K; k0 += 32) {
        // A 16x32 bf16 fragment: grp0 -> K 0-15 ; grp1 -> K 16-31
        v8bf a0 = *(const v8bf*)(arow + k0 + 16 * grp);
        v8bf a1 = *(const v8bf*)(arow + k0 + 16 * grp + 8);
        v16bf a;
#pragma unroll
        for (int i = 0; i < 8; i++) { a[i] = a0[i]; a[i + 8] = a1[i]; }

#pragma unroll
        for (int t = 0; t < 6; t++) {
            // B 32x16 fragment: lanes 0-15 hold K 0-15, lanes 16-31 hold K 16-31
            v16bf b = *(const v16bf*)(&Wt[(t * 16 + mr) * K + k0 + 16 * grp]);
            acc[t] = __builtin_amdgcn_wmma_f32_16x16x32_bf16(
                false, a, false, b, (short)0, acc[t], false, false);
        }
    }

#pragma unroll
    for (int t = 0; t < 6; t++)
#pragma unroll
        for (int r = 0; r < 8; r++)
            C[(long long)(m0 + r + 8 * grp) * HID + t * 16 + mr] = acc[t][r];
}

// ---------------------------------------------------------------------------
// Graph plumbing kernels
// ---------------------------------------------------------------------------
__global__ void fill_f32(float* p, float v, long long n) {
    long long i = (long long)blockIdx.x * blockDim.x + threadIdx.x;
    if (i < n) p[i] = v;
}

__global__ void cvt_bf16(const float* __restrict__ in, bf16_t* __restrict__ out,
                         long long n) {
    long long i = (long long)blockIdx.x * blockDim.x + threadIdx.x;
    if (i < n) out[i] = f2bf(in[i]);
}

__global__ void edge_deg(const long long* __restrict__ dst, float* __restrict__ deg,
                         long long E) {
    long long e = (long long)blockIdx.x * blockDim.x + threadIdx.x;
    if (e < E) atomicAdd(&deg[dst[e]], 1.0f);
}

__global__ void dinv_k(const float* __restrict__ deg, float* __restrict__ dinv, int n) {
    int i = blockIdx.x * blockDim.x + threadIdx.x;
    if (i < n) { float d = deg[i]; dinv[i] = d > 0.f ? rsqrtf(d) : 0.f; }
}

// out[i,f] = h[i,f] * dinv[i]^2   (self-loop contribution; also zero-init)
__global__ void init_out(const float* __restrict__ h, const float* __restrict__ dinv,
                         float* __restrict__ out, long long total) {
    long long i = (long long)blockIdx.x * blockDim.x + threadIdx.x;
    if (i < total) {
        long long node = i / HID;
        float di = dinv[node];
        out[i] = h[i] * di * di;
    }
}

// one wave per edge; 96 feats -> 3 per lane; float atomics resolve in L2
__global__ void edge_agg(const float* __restrict__ h,
                         const long long* __restrict__ src,
                         const long long* __restrict__ dst,
                         const float* __restrict__ dinv,
                         float* __restrict__ out, long long E) {
    long long wid = ((long long)blockIdx.x * blockDim.x + threadIdx.x) >> 5;
    int lane = threadIdx.x & 31;
    if (wid >= E) return;
    long long s = src[wid], d = dst[wid];
    float w = dinv[s] * dinv[d];
    const float* hr = h + s * HID;
    float* orow = out + d * HID;
#pragma unroll
    for (int i = 0; i < 3; i++) {
        int f = lane + 32 * i;
        atomicAdd(&orow[f], hr[f] * w);
    }
}

// bias + relu, emits bf16 (feeds the next WMMA GEMM directly)
__global__ void bias_relu_bf(const float* __restrict__ p, const float* __restrict__ b,
                             bf16_t* __restrict__ out, long long total) {
    long long i = (long long)blockIdx.x * blockDim.x + threadIdx.x;
    if (i < total) {
        float v = p[i] + b[i % HID];
        out[i] = f2bf(v < 0.f ? 0.f : v);
    }
}

// bias (+optional relu), fp32 in-place
__global__ void bias_act(float* __restrict__ p, const float* __restrict__ b,
                         long long total, int relu) {
    long long i = (long long)blockIdx.x * blockDim.x + threadIdx.x;
    if (i < total) {
        float v = p[i] + b[i % HID];
        p[i] = (relu && v < 0.f) ? 0.f : v;
    }
}

__global__ void pool_acc(const float* __restrict__ h, const long long* __restrict__ batch,
                         float* __restrict__ sums, long long total) {
    long long i = (long long)blockIdx.x * blockDim.x + threadIdx.x;
    if (i < total) {
        long long node = i / HID; int f = (int)(i % HID);
        atomicAdd(&sums[batch[node] * HID + f], h[i]);
    }
}

__global__ void cnt_acc(const long long* __restrict__ batch, float* __restrict__ cnt, int n) {
    int i = blockIdx.x * blockDim.x + threadIdx.x;
    if (i < n) atomicAdd(&cnt[batch[i]], 1.0f);
}

// mean pool, emits bf16 for the head GEMM
__global__ void mean_bf(const float* __restrict__ sums, const float* __restrict__ cnt,
                        bf16_t* __restrict__ pooled, long long total) {
    long long i = (long long)blockIdx.x * blockDim.x + threadIdx.x;
    if (i < total) {
        long long g = i / HID;
        pooled[i] = f2bf(sums[i] / fmaxf(cnt[g], 1.0f));
    }
}

// out[g] = dot(T[g,:], Wl2) + bl2    (one 128-thread block per graph)
__global__ void final_dot(const float* __restrict__ T, const float* __restrict__ w2,
                          const float* __restrict__ b2s, float* __restrict__ out) {
    __shared__ float red[128];
    int g = blockIdx.x, j = threadIdx.x;
    float v = 0.f;
    if (j < HID) v = T[(long long)g * HID + j] * w2[j];
    red[j] = v;
    __syncthreads();
    for (int s = 64; s > 0; s >>= 1) {
        if (j < s) red[j] += red[j + s];
        __syncthreads();
    }
    if (j == 0) out[g] = red[0] + b2s[0];
}

// ---------------------------------------------------------------------------
static inline unsigned blocks_for(long long threads) {
    return (unsigned)((threads + 255) / 256);
}

extern "C" void kernel_launch(void* const* d_in, const int* in_sizes, int n_in,
                              void* d_out, int out_size, void* d_ws, size_t ws_size,
                              hipStream_t stream) {
    const float* x   = (const float*)d_in[0];
    const float* W1  = (const float*)d_in[1];
    const float* b1  = (const float*)d_in[2];
    const float* W2  = (const float*)d_in[3];
    const float* b2  = (const float*)d_in[4];
    const float* W3  = (const float*)d_in[5];
    const float* b3  = (const float*)d_in[6];
    const float* Wl1 = (const float*)d_in[7];
    const float* bl1 = (const float*)d_in[8];
    const float* Wl2 = (const float*)d_in[9];
    const float* bl2 = (const float*)d_in[10];
    const long long* eidx  = (const long long*)d_in[11];
    const long long* batch = (const long long*)d_in[12];
    const long long* src = eidx;
    const long long* dst = eidx + NEDGES;
    float* out = (float*)d_out;

    char* w = (char*)d_ws;
    auto carve = [&](size_t bytes) -> char* {
        char* p = w; w += (bytes + 255) & ~(size_t)255; return p;
    };
    bf16_t* xbf    = (bf16_t*)carve((size_t)NNODES * FEAT * 2);  // also reused as abf
    float*  hbuf   = (float*)carve((size_t)NNODES * HID * 4);
    float*  obuf   = (float*)carve((size_t)NNODES * HID * 4);
    float*  deg    = (float*)carve((size_t)NNODES * 4);
    float*  dinv   = (float*)carve((size_t)NNODES * 4);
    float*  sums   = (float*)carve((size_t)NGRAPH * HID * 4);
    float*  cnt    = (float*)carve((size_t)NGRAPH * 4);
    bf16_t* pooled = (bf16_t*)carve((size_t)NGRAPH * HID * 2);
    float*  tmp    = (float*)carve((size_t)NGRAPH * HID * 4);
    bf16_t* abf    = xbf;  // bf16 activations overlay xbf after layer-1 GEMM

    const long long NH = (long long)NNODES * HID;   // 19.2M
    const long long GH = (long long)NGRAPH * HID;
    const long long NF = (long long)NNODES * FEAT;

    // --- normalization: deg (self-loop = 1) + atomic count over edges -> dinv
    fill_f32<<<blocks_for(NNODES), 256, 0, stream>>>(deg, 1.0f, NNODES);
    edge_deg<<<blocks_for(NEDGES), 256, 0, stream>>>(dst, deg, NEDGES);
    dinv_k<<<blocks_for(NNODES), 256, 0, stream>>>(deg, dinv, NNODES);

    // --- bf16 copy of input features
    cvt_bf16<<<blocks_for(NF), 256, 0, stream>>>(x, xbf, NF);

    const unsigned gemmN_blocks = blocks_for((long long)(NNODES / 16) * 32);
    const unsigned gemmG_blocks = blocks_for((long long)(NGRAPH / 16) * 32);
    const unsigned eagg_blocks  = blocks_for((long long)NEDGES * 32);

    // --- layer 1 (K = 128): gemm -> self-loop init -> edge scatter -> bias/relu->bf16
    gemm_wmma_bf16<<<gemmN_blocks, 256, 0, stream>>>(xbf, W1, hbuf, NNODES, FEAT);
    init_out<<<blocks_for(NH), 256, 0, stream>>>(hbuf, dinv, obuf, NH);
    edge_agg<<<eagg_blocks, 256, 0, stream>>>(hbuf, src, dst, dinv, obuf, NEDGES);
    bias_relu_bf<<<blocks_for(NH), 256, 0, stream>>>(obuf, b1, abf, NH);

    // --- layer 2 (K = 96)
    gemm_wmma_bf16<<<gemmN_blocks, 256, 0, stream>>>(abf, W2, hbuf, NNODES, HID);
    init_out<<<blocks_for(NH), 256, 0, stream>>>(hbuf, dinv, obuf, NH);
    edge_agg<<<eagg_blocks, 256, 0, stream>>>(hbuf, src, dst, dinv, obuf, NEDGES);
    bias_relu_bf<<<blocks_for(NH), 256, 0, stream>>>(obuf, b2, abf, NH);

    // --- layer 3 (K = 96), no relu, fp32 output for pooling
    gemm_wmma_bf16<<<gemmN_blocks, 256, 0, stream>>>(abf, W3, hbuf, NNODES, HID);
    init_out<<<blocks_for(NH), 256, 0, stream>>>(hbuf, dinv, obuf, NH);
    edge_agg<<<eagg_blocks, 256, 0, stream>>>(hbuf, src, dst, dinv, obuf, NEDGES);
    bias_act<<<blocks_for(NH), 256, 0, stream>>>(obuf, b3, NH, 0);

    // --- global mean pool (bf16 output feeds head GEMM)
    fill_f32<<<blocks_for(GH), 256, 0, stream>>>(sums, 0.0f, GH);
    fill_f32<<<blocks_for(NGRAPH), 256, 0, stream>>>(cnt, 0.0f, NGRAPH);
    pool_acc<<<blocks_for(NH), 256, 0, stream>>>(obuf, batch, sums, NH);
    cnt_acc<<<blocks_for(NNODES), 256, 0, stream>>>(batch, cnt, NNODES);
    mean_bf<<<blocks_for(GH), 256, 0, stream>>>(sums, cnt, pooled, GH);

    // --- MLP head: WMMA gemm (2048x96 @ 96x96) + bias/relu + final dot
    gemm_wmma_bf16<<<gemmG_blocks, 256, 0, stream>>>(pooled, Wl1, tmp, NGRAPH, HID);
    bias_act<<<blocks_for(GH), 256, 0, stream>>>(tmp, bl1, GH, 1);
    final_dot<<<NGRAPH, 128, 0, stream>>>(tmp, Wl2, bl2, out);
}
